// CriticNetwork_80530636800631
// MI455X (gfx1250) — matrix operand order repaired and checked by reading, not compiled, not run on hardware
//
#include <hip/hip_runtime.h>
#include <hip/hip_bf16.h>

#define BSZ 512
#define NN  250
#define DD  128
#define FFD 512
#define HIDD 256
#define LAY 3
#define BND (BSZ*NN)
#define NPAD 256
#define BNBLK 500   // BND / 256

typedef __bf16 bf16;
typedef __attribute__((ext_vector_type(16))) __bf16 v16bf;
typedef __attribute__((ext_vector_type(8)))  __bf16 v8bf;
typedef __attribute__((ext_vector_type(8)))  float  v8f;

enum { GF_OUT_BF16 = 1, GF_RELU = 2, GF_BIAS = 4, GF_RESID = 8 };

// ---------------------------------------------------------------------------
// WMMA GEMM: C[M,Nn] = A[M,K](bf16,rm) * B[K,Nn](bf16,rm), Nn multiple of 64.
// One wave computes a 16x64 tile: 4 independent f32 accumulators hide the
// WMMA->WMMA RAW hazard; all 8 B-tile loads of a k-step are issued before the
// 4 WMMAs so s_wait_loadcnt can stagger (load j+1..3 overlaps WMMA j).
// FULL=true when the 16 A-rows are all in-bounds (no per-lane predication).
// ---------------------------------------------------------------------------
template<bool FULL>
__device__ __forceinline__ void gemm_core(const bf16* __restrict__ A,
                                          const bf16* __restrict__ B,
                                          int M, int K, int lda, int ldb,
                                          int m0, int n0, int lane, v8f acc[4])
{
    // A operand (16x32 bf16): lane<16 -> row=lane,   K chunks {0..7,16..23}
    //                         lane>=16 -> row=lane-16, K chunks {8..15,24..31}
    const int  arow   = m0 + (lane & 15);
    const int  kofs   = (lane & 16) ? 8 : 0;
    const bool avalid = FULL || (arow < M);
    const bf16* ap = A + (unsigned long long)arow * lda + kofs;
    // B operand (32x16 bf16): lane&15 = N column, elements = K {kofs..+7, 16+kofs..+7}
    const bf16* bp = B + (unsigned long long)kofs * ldb + n0 + (lane & 15);

    #pragma unroll 4
    for (int k0 = 0; k0 < K; k0 += 32) {
        v16bf a;
        if (avalid) {
            v8bf lo = *(const v8bf*)(ap + k0);        // K = k0+kofs    .. +7
            v8bf hi = *(const v8bf*)(ap + k0 + 16);   // K = k0+kofs+16 .. +23
            #pragma unroll
            for (int i = 0; i < 8; ++i) { a[i] = lo[i]; a[i + 8] = hi[i]; }
        } else {
            #pragma unroll
            for (int i = 0; i < 16; ++i) a[i] = (bf16)0.0f;
        }
        const bf16* bq = bp + (unsigned long long)k0 * ldb;
        // Issue all 8 B loads first (distinct regs), then the 4 WMMAs.
        v16bf bt[4];
        #pragma unroll
        for (int j = 0; j < 4; ++j) {
            #pragma unroll
            for (int i = 0; i < 8; ++i) {
                bt[j][i]     = bq[(unsigned long long)i * ldb + j * 16];
                bt[j][i + 8] = bq[(unsigned long long)(16 + i) * ldb + j * 16];
            }
        }
        #pragma unroll
        for (int j = 0; j < 4; ++j) {
            acc[j] = __builtin_amdgcn_wmma_f32_16x16x32_bf16(
                         false, a, false, bt[j], (short)0, acc[j], false, false);
        }
    }
}

__global__ __launch_bounds__(128)
void k_wmma_gemm(const bf16* __restrict__ A, const bf16* __restrict__ B,
                 const float* __restrict__ bias, const float* __restrict__ resid,
                 void* __restrict__ Cout,
                 int M, int Nn, int K, int lda, int ldb, int ldc,
                 unsigned long long sA, unsigned long long sB, unsigned long long sC,
                 int flags)
{
    const int lane = threadIdx.x & 31;
    const int wave = threadIdx.x >> 5;
    const int mt = blockIdx.y * 4 + wave;
    if (mt * 16 >= M) return;
    const unsigned long long bz = blockIdx.z;
    A += bz * sA;
    B += bz * sB;

    const int m0 = mt * 16;
    const int n0 = blockIdx.x * 64;

    v8f acc[4] = {};
    if (m0 + 16 <= M) gemm_core<true >(A, B, M, K, lda, ldb, m0, n0, lane, acc);
    else              gemm_core<false>(A, B, M, K, lda, ldb, m0, n0, lane, acc);

    // C/D layout: VGPR r -> row m0 + r + (lane>=16 ? 8 : 0), col n0 + (lane&15)
    const int lcol  = lane & 15;
    const int crow0 = m0 + ((lane & 16) ? 8 : 0);
    float* Cf = (float*)Cout;
    bf16*  Cb = (bf16*)Cout;
    #pragma unroll
    for (int j = 0; j < 4; ++j) {
        const int ccol = n0 + j * 16 + lcol;
        if (ccol >= Nn) continue;
        float bv = (flags & GF_BIAS) ? bias[ccol] : 0.0f;
        #pragma unroll
        for (int r = 0; r < 8; ++r) {
            int crow = crow0 + r;
            if (crow >= M) continue;
            float v = acc[j][r] + bv;
            unsigned long long ci = bz * sC + (unsigned long long)crow * ldc + ccol;
            if (flags & GF_RESID) v += resid[ci];
            if (flags & GF_RELU)  v = fmaxf(v, 0.0f);
            if (flags & GF_OUT_BF16) Cb[ci] = (bf16)v; else Cf[ci] = v;
        }
    }
}

// ---------------------------------------------------------------------------
// Small elementwise / helper kernels
// ---------------------------------------------------------------------------
__global__ void k_cvt_bf16(const float* __restrict__ s, bf16* __restrict__ d, int n) {
    int t = blockIdx.x * blockDim.x + threadIdx.x;
    if (t < n) d[t] = (bf16)s[t];
}

__global__ void k_addcvt_bf16(const float* __restrict__ a, const float* __restrict__ b,
                              bf16* __restrict__ d, int n) {
    int t = blockIdx.x * blockDim.x + threadIdx.x;
    if (t < n) d[t] = (bf16)(a[t] + b[t]);
}

__global__ void k_zero_bf16(bf16* __restrict__ d, int n) {
    int t = blockIdx.x * blockDim.x + threadIdx.x;
    if (t < n) d[t] = (bf16)0.0f;
}

// pos[b, rec[b,i]-1] = i   (inverse permutation scatter)
__global__ void k_inv_perm(const int* __restrict__ rec, int* __restrict__ pos) {
    int t = blockIdx.x * blockDim.x + threadIdx.x;
    if (t >= BND) return;
    int b = t / NN;
    pos[b * NN + rec[t] - 1] = t % NN;
}

// input_info[b,j,0..6] = [pre.xy, mid.xy, las.xy, demand]
__global__ void k_build_info(const float* __restrict__ loc, const float* __restrict__ dem,
                             const int* __restrict__ rec, const int* __restrict__ pos,
                             float* __restrict__ info) {
    int t = blockIdx.x * blockDim.x + threadIdx.x;
    if (t >= BND) return;
    int b = t / NN, j = t % NN;
    int p = pos[t];
    int pre = rec[b * NN + (p + NN - 1) % NN] - 1;
    int las = rec[b * NN + (p + 1) % NN] - 1;
    const float* lb = loc + (unsigned long long)b * NN * 2;
    float* o = info + (unsigned long long)t * 7;
    o[0] = lb[pre * 2 + 0]; o[1] = lb[pre * 2 + 1];
    o[2] = lb[j * 2 + 0];   o[3] = lb[j * 2 + 1];
    o[4] = lb[las * 2 + 0]; o[5] = lb[las * 2 + 1];
    o[6] = dem[t];
}

// h0 = info @ W_init + b_init + enc[pos]
__global__ void k_embed(const float* __restrict__ info, const float* __restrict__ enc,
                        const float* __restrict__ Wi, const float* __restrict__ bi,
                        const int* __restrict__ pos,
                        float* __restrict__ H, bf16* __restrict__ HB) {
    int t = blockIdx.x * blockDim.x + threadIdx.x;
    if (t >= BND * DD) return;
    int row = t >> 7, d = t & 127;
    int p = pos[row];
    const float* inf = info + (unsigned long long)row * 7;
    float s = bi[d] + enc[p * DD + d];
    #pragma unroll
    for (int c = 0; c < 7; ++c) s += inf[c] * Wi[c * DD + d];
    H[t] = s;
    HB[t] = (bf16)s;
}

// LDS-tiled transpose with zero pad: dst[b,d,np] = (np<NN) ? src[b,np,d] : 0
__global__ __launch_bounds__(256)
void k_transpose_pad(const bf16* __restrict__ src, bf16* __restrict__ dst) {
    __shared__ bf16 tile[32][33];
    const int b  = blockIdx.y;
    const int d0 = (blockIdx.x >> 3) * 32;   // 4 d-tiles
    const int n0 = (blockIdx.x & 7) * 32;    // 8 np-tiles
    const int tx = threadIdx.x & 31;
    const int ty = threadIdx.x >> 5;         // 0..7
    const bf16* sb = src + (unsigned long long)b * NN * DD;
    #pragma unroll
    for (int i = 0; i < 4; ++i) {
        int np = n0 + ty + i * 8;
        tile[ty + i * 8][tx] =
            (np < NN) ? sb[(unsigned long long)np * DD + d0 + tx] : (bf16)0.0f;
    }
    __syncthreads();
    bf16* db = dst + (unsigned long long)b * DD * NPAD;
    #pragma unroll
    for (int i = 0; i < 4; ++i) {
        int d = d0 + ty + i * 8;
        db[(unsigned long long)d * NPAD + n0 + tx] = tile[tx][ty + i * 8];
    }
}

// scaled softmax over 250 valid cols of a 256-wide row; bf16 out, pads=0
__global__ __launch_bounds__(256)
void k_softmax(const float* __restrict__ SC, bf16* __restrict__ P) {
    int row = blockIdx.x;               // 0..BND-1
    int c = threadIdx.x;                // 0..255
    const float scale = 0.08838834764831845f;  // 1/sqrt(128)
    __shared__ float red[256];
    const float* x = SC + (unsigned long long)row * NPAD;
    float v = (c < NN) ? x[c] * scale : -3.4e38f;
    red[c] = v; __syncthreads();
    for (int s = 128; s > 0; s >>= 1) {
        if (c < s) red[c] = fmaxf(red[c], red[c + s]);
        __syncthreads();
    }
    float mx = red[0]; __syncthreads();
    float e = (c < NN) ? __expf(v - mx) : 0.0f;
    red[c] = e; __syncthreads();
    for (int s = 128; s > 0; s >>= 1) {
        if (c < s) red[c] += red[c + s];
        __syncthreads();
    }
    P[(unsigned long long)row * NPAD + c] = (bf16)(e / red[0]);
}

// batchnorm stats, stage 1: coalesced partial sums over 256-row chunks.
__global__ __launch_bounds__(256)
void k_bn_part(const float* __restrict__ X, float* __restrict__ ps,
               float* __restrict__ ps2) {
    const int c = threadIdx.x & 127;
    const int half = threadIdx.x >> 7;
    const int r0 = blockIdx.x * 256;
    float s = 0.f, s2 = 0.f;
    for (int i = half; i < 256; i += 2) {
        float v = X[(unsigned long long)(r0 + i) * DD + c];
        s += v; s2 += v * v;
    }
    __shared__ float sh[256], sh2[256];
    sh[threadIdx.x] = s; sh2[threadIdx.x] = s2;
    __syncthreads();
    if (half == 0) {
        ps [blockIdx.x * DD + c] = sh [c] + sh [c + 128];
        ps2[blockIdx.x * DD + c] = sh2[c] + sh2[c + 128];
    }
}

// batchnorm stats, stage 2: deterministic final reduce (biased variance)
__global__ __launch_bounds__(128)
void k_bn_finish(const float* __restrict__ ps, const float* __restrict__ ps2,
                 float* __restrict__ mean, float* __restrict__ rstd) {
    int c = threadIdx.x;
    float s = 0.f, s2 = 0.f;
    for (int g = 0; g < BNBLK; ++g) { s += ps[g * DD + c]; s2 += ps2[g * DD + c]; }
    float m = s / (float)BND;
    float var = s2 / (float)BND - m * m;
    mean[c] = m;
    rstd[c] = rsqrtf(var + 1e-5f);
}

__global__ void k_bn_apply(const float* __restrict__ X, const float* __restrict__ mean,
                           const float* __restrict__ rstd, const float* __restrict__ w,
                           const float* __restrict__ bb,
                           float* __restrict__ H, bf16* __restrict__ HB) {
    int t = blockIdx.x * blockDim.x + threadIdx.x;
    if (t >= BND * DD) return;
    int c = t & 127;
    float y = (X[t] - mean[c]) * rstd[c] * w[c] + bb[c];
    H[t] = y;
    HB[t] = (bf16)y;
}

// graph_embed[b,d] = mean_n H[b,:,d]  (bf16 out, feeds head GEMM)
__global__ void k_mean_rows(const float* __restrict__ H, bf16* __restrict__ GE) {
    int t = blockIdx.x * blockDim.x + threadIdx.x;
    if (t >= BSZ * DD) return;
    int b = t >> 7, d = t & 127;
    float s = 0.f;
    const float* p = H + (unsigned long long)b * NN * DD + d;
    for (int j = 0; j < NN; ++j) s += p[(unsigned long long)j * DD];
    GE[t] = (bf16)(s / (float)NN);
}

// out[b] = hid[b,:] . v2_w + v2_b
__global__ void k_head_out(const bf16* __restrict__ HID, const float* __restrict__ v2w,
                           const float* __restrict__ v2b, float* __restrict__ out) {
    int b = blockIdx.x * blockDim.x + threadIdx.x;
    if (b >= BSZ) return;
    float s = v2b[0];
    const bf16* h = HID + (unsigned long long)b * HIDD;
    for (int i = 0; i < HIDD; ++i) s += (float)h[i] * v2w[i];
    out[b] = s;
}

// ---------------------------------------------------------------------------
// Host side
// ---------------------------------------------------------------------------
static inline void gemm(hipStream_t st, const bf16* A, const bf16* B, const float* bias,
                        const float* resid, void* C, int M, int Nn, int K,
                        int lda, int ldb, int ldc,
                        unsigned long long sA, unsigned long long sB, unsigned long long sC,
                        int batches, int flags) {
    dim3 grid(Nn / 64, (((M + 15) / 16) + 3) / 4, batches);
    k_wmma_gemm<<<grid, dim3(128), 0, st>>>(A, B, bias, resid, C,
                                            M, Nn, K, lda, ldb, ldc, sA, sB, sC, flags);
}

static inline void cvt(hipStream_t st, const float* s, bf16* d, int n) {
    k_cvt_bf16<<<(n + 255) / 256, 256, 0, st>>>(s, d, n);
}

extern "C" void kernel_launch(void* const* d_in, const int* in_sizes, int n_in,
                              void* d_out, int out_size, void* d_ws, size_t ws_size,
                              hipStream_t stream) {
    (void)in_sizes; (void)n_in; (void)out_size; (void)ws_size;
    const float* loc    = (const float*)d_in[0];
    const float* demand = (const float*)d_in[1];
    const float* enc    = (const float*)d_in[2];
    const float* W_init = (const float*)d_in[3];
    const float* b_init = (const float*)d_in[4];
    const float* Wq     = (const float*)d_in[5];
    const float* Wk     = (const float*)d_in[6];
    const float* Wv     = (const float*)d_in[7];
    const float* Wo     = (const float*)d_in[8];
    const float* bn1_w  = (const float*)d_in[9];
    const float* bn1_b  = (const float*)d_in[10];
    const float* ff_w1  = (const float*)d_in[11];
    const float* ff_b1  = (const float*)d_in[12];
    const float* ff_w2  = (const float*)d_in[13];
    const float* ff_b2  = (const float*)d_in[14];
    const float* bn2_w  = (const float*)d_in[15];
    const float* bn2_b  = (const float*)d_in[16];
    const float* Wg     = (const float*)d_in[17];
    const float* Wn     = (const float*)d_in[18];
    const float* v1_w   = (const float*)d_in[19];
    const float* v1_b   = (const float*)d_in[20];
    const float* v2_w   = (const float*)d_in[21];
    const float* v2_b   = (const float*)d_in[22];
    const int*   rec    = (const int*)d_in[23];
    float* out = (float*)d_out;

    char* ws = (char*)d_ws;
    size_t off = 0;
    auto alloc = [&](size_t bytes) -> void* {
        off = (off + 255) & ~(size_t)255;
        void* p = ws + off;
        off += bytes;
        return p;
    };

    int*   pos   = (int*)  alloc((size_t)BND * 4);
    float* info  = (float*)alloc((size_t)BND * 7 * 4);
    float* H     = (float*)alloc((size_t)BND * DD * 4);
    bf16*  HB    = (bf16*) alloc((size_t)BND * DD * 2);
    float* TMP   = (float*)alloc((size_t)BND * DD * 4);
    bf16*  QB    = (bf16*) alloc((size_t)BND * DD * 2);   // reused as attn output
    bf16*  KB    = (bf16*) alloc((size_t)BND * DD * 2);
    bf16*  VBP   = (bf16*) alloc((size_t)BSZ * NPAD * DD * 2);
    bf16*  KT    = (bf16*) alloc((size_t)BSZ * DD * NPAD * 2);
    float* SC    = (float*)alloc((size_t)BND * NPAD * 4); // reused as FF1 (bf16)
    bf16*  PBF   = (bf16*) alloc((size_t)BND * NPAD * 2);
    bf16*  wqb   = (bf16*) alloc((size_t)LAY * DD * DD * 2);
    bf16*  wkb   = (bf16*) alloc((size_t)LAY * DD * DD * 2);
    bf16*  wvb   = (bf16*) alloc((size_t)LAY * DD * DD * 2);
    bf16*  wob   = (bf16*) alloc((size_t)LAY * DD * DD * 2);
    bf16*  f1b   = (bf16*) alloc((size_t)LAY * DD * FFD * 2);
    bf16*  f2b   = (bf16*) alloc((size_t)LAY * FFD * DD * 2);
    bf16*  wcb   = (bf16*) alloc((size_t)DD * DD * 2);
    bf16*  v1b   = (bf16*) alloc((size_t)DD * HIDD * 2);
    float* psum  = (float*)alloc((size_t)BNBLK * DD * 4);
    float* psum2 = (float*)alloc((size_t)BNBLK * DD * 4);
    float* bmean = (float*)alloc(DD * 4);
    float* brstd = (float*)alloc(DD * 4);
    bf16*  GEB   = (bf16*) alloc((size_t)BSZ * DD * 2);
    bf16*  FUSB  = (bf16*) alloc((size_t)BSZ * DD * 2);
    bf16*  HIDB  = (bf16*) alloc((size_t)BSZ * HIDD * 2);
    bf16*  ATTB  = QB;                    // attn output aliases Q (Q dead by then)
    bf16*  FF1B  = (bf16*)SC;             // FF1 aliases scores (dead after softmax)

    // ---- weight conversions (deterministic each call) ----
    cvt(stream, Wq, wqb, LAY * DD * DD);
    cvt(stream, Wk, wkb, LAY * DD * DD);
    cvt(stream, Wv, wvb, LAY * DD * DD);
    cvt(stream, Wo, wob, LAY * DD * DD);
    cvt(stream, ff_w1, f1b, LAY * DD * FFD);
    cvt(stream, ff_w2, f2b, LAY * FFD * DD);
    cvt(stream, v1_w, v1b, DD * HIDD);
    k_addcvt_bf16<<<(DD * DD + 255) / 256, 256, 0, stream>>>(Wg, Wn, wcb, DD * DD);

    // ---- embedding ----
    k_inv_perm<<<(BND + 255) / 256, 256, 0, stream>>>(rec, pos);
    k_build_info<<<(BND + 255) / 256, 256, 0, stream>>>(loc, demand, rec, pos, info);
    k_embed<<<(BND * DD + 255) / 256, 256, 0, stream>>>(info, enc, W_init, b_init, pos, H, HB);

    // V is written batched into a 256-row padded layout; pad rows must be zero.
    k_zero_bf16<<<(BSZ * NPAD * DD + 255) / 256, 256, 0, stream>>>(VBP, BSZ * NPAD * DD);

    // ---- encoder layers ----
    for (int l = 0; l < LAY; ++l) {
        const bf16* wq_l = wqb + (size_t)l * DD * DD;
        const bf16* wk_l = wkb + (size_t)l * DD * DD;
        const bf16* wv_l = wvb + (size_t)l * DD * DD;
        const bf16* wo_l = wob + (size_t)l * DD * DD;
        const bf16* f1_l = f1b + (size_t)l * DD * FFD;
        const bf16* f2_l = f2b + (size_t)l * FFD * DD;

        // Q, K projections (flat M = 128000)
        gemm(stream, HB, wq_l, nullptr, nullptr, QB, BND, DD, DD, DD, DD, DD,
             0, 0, 0, 1, GF_OUT_BF16);
        gemm(stream, HB, wk_l, nullptr, nullptr, KB, BND, DD, DD, DD, DD, DD,
             0, 0, 0, 1, GF_OUT_BF16);
        // V projection, batched into padded (BS, 256, 128)
        gemm(stream, HB, wv_l, nullptr, nullptr, VBP, NN, DD, DD, DD, DD, DD,
             (unsigned long long)NN * DD, 0, (unsigned long long)NPAD * DD,
             BSZ, GF_OUT_BF16);
        // K^T with zero pad (LDS tiled)
        k_transpose_pad<<<dim3(32, BSZ), 256, 0, stream>>>(KB, KT);
        // scores = Q @ K^T, batched (M=250, N=256, K=128)
        gemm(stream, QB, KT, nullptr, nullptr, SC, NN, NPAD, DD, DD, NPAD, NPAD,
             (unsigned long long)NN * DD, (unsigned long long)DD * NPAD,
             (unsigned long long)NN * NPAD, BSZ, 0);
        // softmax -> bf16 probs (padded cols zeroed)
        k_softmax<<<BND, 256, 0, stream>>>(SC, PBF);
        // attn @ V, batched (M=250, N=128, K=256) -> bf16 into ATTB (= QB region)
        gemm(stream, PBF, VBP, nullptr, nullptr, ATTB, NN, DD, NPAD, NPAD, DD, DD,
             (unsigned long long)NN * NPAD, (unsigned long long)NPAD * DD,
             (unsigned long long)NN * DD, BSZ, GF_OUT_BF16);
        // mha @ Wo + residual(h) -> TMP (f32)
        gemm(stream, ATTB, wo_l, nullptr, H, TMP, BND, DD, DD, DD, DD, DD,
             0, 0, 0, 1, GF_RESID);
        // batchnorm 1
        k_bn_part<<<BNBLK, 256, 0, stream>>>(TMP, psum, psum2);
        k_bn_finish<<<1, 128, 0, stream>>>(psum, psum2, bmean, brstd);
        k_bn_apply<<<(BND * DD + 255) / 256, 256, 0, stream>>>(
            TMP, bmean, brstd, bn1_w + l * DD, bn1_b + l * DD, H, HB);
        // FF1: relu(h @ w1 + b1) -> bf16
        gemm(stream, HB, f1_l, ff_b1 + l * FFD, nullptr, FF1B, BND, FFD, DD,
             DD, FFD, FFD, 0, 0, 0, 1, GF_OUT_BF16 | GF_RELU | GF_BIAS);
        // FF2: ff1 @ w2 + b2 + residual(h) -> TMP (f32)
        gemm(stream, FF1B, f2_l, ff_b2 + l * DD, H, TMP, BND, DD, FFD,
             FFD, DD, DD, 0, 0, 0, 1, GF_BIAS | GF_RESID);
        // batchnorm 2
        k_bn_part<<<BNBLK, 256, 0, stream>>>(TMP, psum, psum2);
        k_bn_finish<<<1, 128, 0, stream>>>(psum, psum2, bmean, brstd);
        k_bn_apply<<<(BND * DD + 255) / 256, 256, 0, stream>>>(
            TMP, bmean, brstd, bn2_w + l * DD, bn2_b + l * DD, H, HB);
    }

    // ---- value head: fusion = graph_embed @ (Wg + Wn) ----
    k_mean_rows<<<(BSZ * DD + 255) / 256, 256, 0, stream>>>(H, GEB);
    gemm(stream, GEB, wcb, nullptr, nullptr, FUSB, BSZ, DD, DD, DD, DD, DD,
         0, 0, 0, 1, GF_OUT_BF16);
    gemm(stream, FUSB, v1b, v1_b, nullptr, HIDB, BSZ, HIDD, DD, DD, HIDD, HIDD,
         0, 0, 0, 1, GF_OUT_BF16 | GF_RELU | GF_BIAS);
    k_head_out<<<(BSZ + 255) / 256, 256, 0, stream>>>(HIDB, v2_w, v2_b, out);
}
